// IncrementalDecoderLayer_89275190215316
// MI455X (gfx1250) — matrix operand order repaired and checked by reading, not compiled, not run on hardware
//
#include <hip/hip_runtime.h>

// ---------------------------------------------------------------- constants
constexpr int Bn  = 1024;
constexpr int Dm  = 1024;
constexpr int Hh  = 16;
constexpr int Ww  = 256;
constexpr int Ll  = 16;
constexpr int DFF = 4096;
constexpr int DH  = 64;      // Dm / Hh

typedef __attribute__((ext_vector_type(16))) __bf16 v16bf;
typedef __attribute__((ext_vector_type(8)))  float  v8f;

union FragBF {
    v16bf v;
    uint4 q[2];
};

static __device__ __forceinline__ unsigned short f32_to_bf16(float f) {
    union { float f; unsigned int u; } a; a.f = f;
    unsigned int r = a.u + 0x7FFFu + ((a.u >> 16) & 1u);   // round-nearest-even
    return (unsigned short)(r >> 16);
}

// --------------------------------------------------- weight convert+transpose
// in:  f32 [K, N] row-major   out: bf16 [N, K] row-major
__global__ void tconv_kernel(const float* __restrict__ in,
                             unsigned short* __restrict__ out,
                             int K, int N) {
    size_t idx = (size_t)blockIdx.x * 256 + threadIdx.x;
    size_t tot = (size_t)K * N;
    if (idx >= tot) return;
    int k = (int)(idx / N);
    int n = (int)(idx % N);
    out[(size_t)n * K + k] = f32_to_bf16(in[idx]);
}

// ---------------------------------------------------------------- RMSNorm
__global__ __launch_bounds__(256) void rmsnorm_kernel(
        const float* __restrict__ x, const float* __restrict__ w,
        float* __restrict__ outF, unsigned short* __restrict__ outB) {
    __shared__ float red[256];
    int b = blockIdx.x, tid = threadIdx.x;
    const float* xr = x + (size_t)b * Dm;
    float v[4]; float s = 0.f;
#pragma unroll
    for (int i = 0; i < 4; ++i) { v[i] = xr[tid + i * 256]; s += v[i] * v[i]; }
    red[tid] = s; __syncthreads();
    for (int st = 128; st; st >>= 1) {
        if (tid < st) red[tid] += red[tid + st];
        __syncthreads();
    }
    float r = rsqrtf(red[0] * (1.0f / Dm) + 1e-6f);
#pragma unroll
    for (int i = 0; i < 4; ++i) {
        int idx = tid + i * 256;
        float val = v[i] * r * w[idx];
        if (outF) outF[(size_t)b * Dm + idx] = val;
        if (outB) outB[(size_t)b * Dm + idx] = f32_to_bf16(val);
    }
}

// ---------------------------------------------------------------- WMMA GEMM
// helpers: all fragment indices are compile-time constants after inlining,
// so everything stays in VGPRs (no runtime-indexed arrays -> no scratch).
static __device__ __forceinline__ void load_frags(
        FragBF (&aF)[2], FragBF (&bF)[4],
        const unsigned short* aptr0, const unsigned short* aptr1,
        const unsigned short* bptr0, const unsigned short* bptr1,
        const unsigned short* bptr2, const unsigned short* bptr3, int ko) {
    aF[0].q[0] = *(const uint4*)(aptr0 + ko);
    aF[0].q[1] = *(const uint4*)(aptr0 + ko + 8);
    aF[1].q[0] = *(const uint4*)(aptr1 + ko);
    aF[1].q[1] = *(const uint4*)(aptr1 + ko + 8);
    bF[0].q[0] = *(const uint4*)(bptr0 + ko);
    bF[0].q[1] = *(const uint4*)(bptr0 + ko + 8);
    bF[1].q[0] = *(const uint4*)(bptr1 + ko);
    bF[1].q[1] = *(const uint4*)(bptr1 + ko + 8);
    bF[2].q[0] = *(const uint4*)(bptr2 + ko);
    bF[2].q[1] = *(const uint4*)(bptr2 + ko + 8);
    bF[3].q[0] = *(const uint4*)(bptr3 + ko);
    bF[3].q[1] = *(const uint4*)(bptr3 + ko + 8);
}

static __device__ __forceinline__ void do_wmma(
        v8f (&acc)[8], const FragBF (&aF)[2], const FragBF (&bF)[4]) {
#pragma unroll
    for (int t = 0; t < 4; ++t) {
        acc[t] = __builtin_amdgcn_wmma_f32_16x16x32_bf16(
            false, aF[0].v, false, bF[t].v, (short)0, acc[t], false, false);
        acc[4 + t] = __builtin_amdgcn_wmma_f32_16x16x32_bf16(
            false, aF[1].v, false, bF[t].v, (short)0, acc[4 + t], false, false);
    }
}

// C[M,N] = act( A[M,K](bf16) @ Bt[N,K]^T(bf16) + bias + resid )
// one wave -> 32 rows x 64 cols (8 x 16x16 tiles); two named register buffer
// sets, k-loop unrolled by 2 with last step peeled (K % 64 == 0, K >= 128),
// so prefetch loads overlap each 8-WMMA compute phase without spills.
__global__ __launch_bounds__(256) void gemm_bf16_kernel(
        const unsigned short* __restrict__ A,    // [M,K] bf16
        const unsigned short* __restrict__ Bt,   // [N,K] bf16 (transposed weights)
        const float* __restrict__ bias,          // [N]
        const float* __restrict__ resid,         // [M,N] or null
        float* __restrict__ outF,                // [M,N] or null
        unsigned short* __restrict__ outB,       // [M,N] or null
        int N, int K, int act) {
    const int lane = threadIdx.x & 31;
    const int wave = threadIdx.x >> 5;
    const int row0 = blockIdx.y * 256 + wave * 32;
    const int col0 = blockIdx.x * 64;
    const int m    = lane & 15;
    const int kbA  = (lane < 16) ? 0 : 16;   // A frag: per-lane K base
    const int kbB  = (lane < 16) ? 0 : 16;   // B frag: per-lane K base

    const unsigned short* aptr0 = A + (size_t)(row0 + m) * K + kbA;
    const unsigned short* aptr1 = A + (size_t)(row0 + 16 + m) * K + kbA;
    const unsigned short* bptr0 = Bt + (size_t)(col0 + 0  + m) * K + kbB;
    const unsigned short* bptr1 = Bt + (size_t)(col0 + 16 + m) * K + kbB;
    const unsigned short* bptr2 = Bt + (size_t)(col0 + 32 + m) * K + kbB;
    const unsigned short* bptr3 = Bt + (size_t)(col0 + 48 + m) * K + kbB;

    v8f acc[8] = {};                 // [row-tile 0..1][col-tile 0..3]
    FragBF aA[2], bA[4];             // buffer set A
    FragBF aB[2], bB[4];             // buffer set B

    load_frags(aA, bA, aptr0, aptr1, bptr0, bptr1, bptr2, bptr3, 0);

    int kb = 0;
    for (; kb + 64 < K; kb += 64) {
        load_frags(aB, bB, aptr0, aptr1, bptr0, bptr1, bptr2, bptr3, kb + 32);
        do_wmma(acc, aA, bA);
        load_frags(aA, bA, aptr0, aptr1, bptr0, bptr1, bptr2, bptr3, kb + 64);
        do_wmma(acc, aB, bB);
    }
    // tail: kb == K - 64
    load_frags(aB, bB, aptr0, aptr1, bptr0, bptr1, bptr2, bptr3, kb + 32);
    do_wmma(acc, aA, bA);
    do_wmma(acc, aB, bB);

    const int nbcol = lane & 15;
    const int mofs  = (lane < 16) ? 0 : 8;
#pragma unroll
    for (int rt = 0; rt < 2; ++rt) {
#pragma unroll
        for (int t = 0; t < 4; ++t) {
            int col = col0 + t * 16 + nbcol;
            float bi = bias[col];
#pragma unroll
            for (int r = 0; r < 8; ++r) {
                int row = row0 + rt * 16 + mofs + r;
                float v = acc[rt * 4 + t][r] + bi;
                if (resid) v += resid[(size_t)row * N + col];
                if (act == 1) v = 0.5f * v * (1.f + erff(v * 0.70710678f));  // exact GELU
                if (outF) outF[(size_t)row * N + col] = v;
                if (outB) outB[(size_t)row * N + col] = f32_to_bf16(v);
            }
        }
    }
}

// ---------------------------------------------------------------- gate GEMV
// g[b,h] = sigmoid( h_row . Wg[:,h] + bg[h] )
__global__ __launch_bounds__(128) void gate_kernel(
        const float* __restrict__ h, const float* __restrict__ Wg,
        const float* __restrict__ bg, float* __restrict__ g) {
    __shared__ float part[8][16];
    int b = blockIdx.x, t = threadIdx.x;
    int hh = t & 15, c = t >> 4;
    const float* hr = h + (size_t)b * Dm;
    float acc = 0.f;
    for (int d = c * 128; d < c * 128 + 128; ++d)
        acc += hr[d] * Wg[d * Hh + hh];
    part[c][hh] = acc; __syncthreads();
    if (t < 16) {
        float s = bg[t];
#pragma unroll
        for (int c2 = 0; c2 < 8; ++c2) s += part[c2][t];
        g[b * Hh + t] = 1.f / (1.f + __expf(-s));
    }
}

// ---------------------------------------------------------------- attention
// one block per batch row; 8 waves, 2 heads/wave; memory-bound fp32 streaming
__global__ __launch_bounds__(256) void attn_kernel(
        const float* __restrict__ q,        // [B,D]
        const float* __restrict__ k_cache,  // [B,W,D]
        const float* __restrict__ v_cache,  // [B,W,D]
        const float* __restrict__ k_new,    // [B,D]
        const float* __restrict__ v_new,    // [B,D]
        const float* __restrict__ lat_k,    // [B,L,D]
        const float* __restrict__ lat_v,    // [B,L,D]
        const float* __restrict__ g,        // [B,H]
        unsigned short* __restrict__ attn_bf) {  // [B,D] bf16
    __shared__ float q_s[Dm];
    __shared__ float s_loc[Hh][Ww + 1];
    __shared__ float s_lat[Hh][Ll];

    const int b    = blockIdx.x;
    const int tid  = threadIdx.x;
    const int lane = tid & 31;
    const int wave = tid >> 5;
    const float scale = 0.125f;   // 1/sqrt(64)

    for (int i = tid; i < Dm; i += 256) q_s[i] = q[(size_t)b * Dm + i];
    __syncthreads();

    // ---- phase A: scores ----
    for (int i = 0; i < 2; ++i) {
        int hh = wave * 2 + i;
        const float* qh = &q_s[hh * DH];
        for (int base = 0; base < Ww; base += 32) {
            int key = base + lane;
            const float* kr = k_cache + ((size_t)b * Ww + key) * Dm + hh * DH;
            float acc = 0.f;
#pragma unroll
            for (int d = 0; d < DH; d += 4) {
                float4 kv = *(const float4*)(kr + d);
                acc += qh[d] * kv.x + qh[d + 1] * kv.y +
                       qh[d + 2] * kv.z + qh[d + 3] * kv.w;
            }
            s_loc[hh][key] = acc * scale;
        }
        if (lane == 0) {
            const float* kr = k_new + (size_t)b * Dm + hh * DH;
            float acc = 0.f;
#pragma unroll
            for (int d = 0; d < DH; ++d) acc += qh[d] * kr[d];
            s_loc[hh][Ww] = acc * scale;
        }
        if (lane < Ll) {
            const float* kr = lat_k + ((size_t)b * Ll + lane) * Dm + hh * DH;
            float acc = 0.f;
#pragma unroll
            for (int d = 0; d < DH; ++d) acc += qh[d] * kr[d];
            s_lat[hh][lane] = acc * scale;
        }
    }
    __syncthreads();

    // ---- phase B/C: softmax + PV + gate ----
    for (int i = 0; i < 2; ++i) {
        int hh = wave * 2 + i;
        // local softmax (257 keys)
        float mx = -1e30f;
        for (int key = lane; key <= Ww; key += 32) mx = fmaxf(mx, s_loc[hh][key]);
        for (int o = 16; o; o >>= 1) mx = fmaxf(mx, __shfl_xor(mx, o, 32));
        float sum = 0.f;
        for (int key = lane; key <= Ww; key += 32) {
            float p = __expf(s_loc[hh][key] - mx);
            s_loc[hh][key] = p; sum += p;
        }
        for (int o = 16; o; o >>= 1) sum += __shfl_xor(sum, o, 32);
        float inv_loc = 1.f / sum;
        // latent softmax (16 keys)
        float sl = (lane < Ll) ? s_lat[hh][lane] : -1e30f;
        float mxl = sl;
        for (int o = 16; o; o >>= 1) mxl = fmaxf(mxl, __shfl_xor(mxl, o, 32));
        float pl = (lane < Ll) ? __expf(sl - mxl) : 0.f;
        if (lane < Ll) s_lat[hh][lane] = pl;
        float suml = pl;
        for (int o = 16; o; o >>= 1) suml += __shfl_xor(suml, o, 32);
        float inv_lat = 1.f / suml;
        __syncthreads();   // make this wave's p-values LDS-visible (uniform)

        // PV: lanes over dims (2 dims/lane), coalesced float2 streams
        int d0 = hh * DH + lane * 2;
        float a0 = 0.f, a1 = 0.f;
        for (int key = 0; key < Ww; ++key) {
            float p = s_loc[hh][key];
            float2 vv = *(const float2*)(v_cache + ((size_t)b * Ww + key) * Dm + d0);
            a0 += p * vv.x; a1 += p * vv.y;
        }
        {
            float p = s_loc[hh][Ww];
            float2 vv = *(const float2*)(v_new + (size_t)b * Dm + d0);
            a0 += p * vv.x; a1 += p * vv.y;
        }
        a0 *= inv_loc; a1 *= inv_loc;
        float l0 = 0.f, l1 = 0.f;
#pragma unroll
        for (int key = 0; key < Ll; ++key) {
            float p = s_lat[hh][key];
            float2 vv = *(const float2*)(lat_v + ((size_t)b * Ll + key) * Dm + d0);
            l0 += p * vv.x; l1 += p * vv.y;
        }
        l0 *= inv_lat; l1 *= inv_lat;
        float gg = g[b * Hh + hh];
        float o0 = gg * a0 + (1.f - gg) * l0;
        float o1 = gg * a1 + (1.f - gg) * l1;
        unsigned int packed = (unsigned int)f32_to_bf16(o0) |
                              ((unsigned int)f32_to_bf16(o1) << 16);
        *(unsigned int*)(attn_bf + (size_t)b * Dm + d0) = packed;
    }
}

// ---------------------------------------------------------------- launcher
extern "C" void kernel_launch(void* const* d_in, const int* in_sizes, int n_in,
                              void* d_out, int out_size, void* d_ws, size_t ws_size,
                              hipStream_t stream) {
    (void)in_sizes; (void)n_in; (void)out_size; (void)ws_size;
    const float* x_t     = (const float*)d_in[0];
    const float* k_cache = (const float*)d_in[1];
    const float* v_cache = (const float*)d_in[2];
    const float* lat_k   = (const float*)d_in[3];
    const float* lat_v   = (const float*)d_in[4];
    const float* norm1_w = (const float*)d_in[5];
    const float* norm2_w = (const float*)d_in[6];
    const float* Wq = (const float*)d_in[7];  const float* bq = (const float*)d_in[8];
    const float* Wk = (const float*)d_in[9];  const float* bk = (const float*)d_in[10];
    const float* Wv = (const float*)d_in[11]; const float* bv = (const float*)d_in[12];
    const float* Wo = (const float*)d_in[13]; const float* bo = (const float*)d_in[14];
    const float* Wg = (const float*)d_in[15]; const float* bg = (const float*)d_in[16];
    const float* w1 = (const float*)d_in[17]; const float* b1 = (const float*)d_in[18];
    const float* w2 = (const float*)d_in[19]; const float* b2 = (const float*)d_in[20];
    float* out = (float*)d_out;

    char* wsp = (char*)d_ws;
    auto alloc = [&](size_t bytes) {
        char* p = wsp; wsp += (bytes + 255) & ~(size_t)255; return p;
    };
    unsigned short* WqT  = (unsigned short*)alloc((size_t)Dm * Dm * 2);
    unsigned short* WkT  = (unsigned short*)alloc((size_t)Dm * Dm * 2);
    unsigned short* WvT  = (unsigned short*)alloc((size_t)Dm * Dm * 2);
    unsigned short* WoT  = (unsigned short*)alloc((size_t)Dm * Dm * 2);
    unsigned short* w1T  = (unsigned short*)alloc((size_t)Dm * DFF * 2);
    unsigned short* w2T  = (unsigned short*)alloc((size_t)DFF * Dm * 2);
    float*          h_f  = (float*)alloc((size_t)Bn * Dm * 4);
    unsigned short* h_b  = (unsigned short*)alloc((size_t)Bn * Dm * 2);
    float*          q_f  = (float*)alloc((size_t)Bn * Dm * 4);
    float*          kn_f = (float*)alloc((size_t)Bn * Dm * 4);
    float*          vn_f = (float*)alloc((size_t)Bn * Dm * 4);
    float*          g_f  = (float*)alloc((size_t)Bn * Hh * 4);
    unsigned short* at_b = (unsigned short*)alloc((size_t)Bn * Dm * 2);
    float*          x_f  = (float*)alloc((size_t)Bn * Dm * 4);
    unsigned short* h2_b = (unsigned short*)alloc((size_t)Bn * Dm * 2);
    unsigned short* ff_b = (unsigned short*)alloc((size_t)Bn * DFF * 2);

    // weight convert + transpose to bf16 [N,K]
    int nDD = (Dm * Dm + 255) / 256, nDF = (Dm * DFF + 255) / 256;
    tconv_kernel<<<nDD, 256, 0, stream>>>(Wq, WqT, Dm, Dm);
    tconv_kernel<<<nDD, 256, 0, stream>>>(Wk, WkT, Dm, Dm);
    tconv_kernel<<<nDD, 256, 0, stream>>>(Wv, WvT, Dm, Dm);
    tconv_kernel<<<nDD, 256, 0, stream>>>(Wo, WoT, Dm, Dm);
    tconv_kernel<<<nDF, 256, 0, stream>>>(w1, w1T, Dm, DFF);
    tconv_kernel<<<nDF, 256, 0, stream>>>(w2, w2T, DFF, Dm);

    // norm1
    rmsnorm_kernel<<<Bn, 256, 0, stream>>>(x_t, norm1_w, h_f, h_b);

    // QKV projections (WMMA bf16)
    dim3 gemmGridDD(Dm / 64, Bn / 256);
    gemm_bf16_kernel<<<gemmGridDD, 256, 0, stream>>>(h_b, WqT, bq, nullptr, q_f,  nullptr, Dm, Dm, 0);
    gemm_bf16_kernel<<<gemmGridDD, 256, 0, stream>>>(h_b, WkT, bk, nullptr, kn_f, nullptr, Dm, Dm, 0);
    gemm_bf16_kernel<<<gemmGridDD, 256, 0, stream>>>(h_b, WvT, bv, nullptr, vn_f, nullptr, Dm, Dm, 0);

    // gate
    gate_kernel<<<Bn, 128, 0, stream>>>(h_f, Wg, bg, g_f);

    // attention (memory-bound: streams 2.15 GB of KV)
    attn_kernel<<<Bn, 256, 0, stream>>>(q_f, k_cache, v_cache, kn_f, vn_f,
                                        lat_k, lat_v, g_f, at_b);

    // output projection + residual -> x
    gemm_bf16_kernel<<<gemmGridDD, 256, 0, stream>>>(at_b, WoT, bo, x_t, x_f, nullptr, Dm, Dm, 0);

    // norm2
    rmsnorm_kernel<<<Bn, 256, 0, stream>>>(x_f, norm2_w, nullptr, h2_b);

    // FFN1 (+GELU) and FFN2 (+residual) -> out
    dim3 gemmGridDF(DFF / 64, Bn / 256);
    gemm_bf16_kernel<<<gemmGridDF, 256, 0, stream>>>(h2_b, w1T, b1, nullptr, nullptr, ff_b, DFF, Dm, 1);
    gemm_bf16_kernel<<<gemmGridDD, 256, 0, stream>>>(ff_b, w2T, b2, x_f, out, nullptr, Dm, DFF, 0);
}
